// EDKT_11321533792494
// MI455X (gfx1250) — compile-verified, matrix operand order checked by reading, not run on hardware
//
#include <hip/hip_runtime.h>
#include <hip/hip_bf16.h>
#include <stdint.h>

// Problem constants (from reference)
#define BB   16
#define SS   128
#define EE   128
#define TTT  20
#define LLL  128
#define OUTD 100
#define ERRTH 50
#define ROWS (BB*SS)          // 2048

typedef __attribute__((ext_vector_type(16))) __bf16 v16bf;
typedef __attribute__((ext_vector_type(8)))  float  v8f;
typedef __attribute__((ext_vector_type(4)))  unsigned int u32x4;
typedef __attribute__((ext_vector_type(8)))  int i32x8;
typedef __attribute__((ext_vector_type(4)))  int i32x4;

union Frag {
  v16bf v;
  unsigned short s[16];
  uint4 q[2];
};

__device__ __forceinline__ unsigned short f2bf(float f){
  unsigned int u = __float_as_uint(f);
  u += 0x7FFFu + ((u >> 16) & 1u);      // round-to-nearest-even
  return (unsigned short)(u >> 16);
}
__device__ __forceinline__ float sigf(float x){ return 1.f/(1.f+__expf(-x)); }

__device__ __forceinline__ v8f wmma_bf16(const Frag& a, const Frag& b, v8f c){
  // v_wmma_f32_16x16x32_bf16  (neg_a, A, neg_b, B, c_mod, C, reuse_a, reuse_b)
  return __builtin_amdgcn_wmma_f32_16x16x32_bf16(false, a.v, false, b.v, (short)0, c, false, false);
}

// Load a 16x32 bf16 A-fragment from a row-major bf16 matrix.
// Row-major 16-bit data maps onto the WMMA A layout with two B128 loads per
// lane (ISA 10.9):
//   frag elems 0..7  = K = kbase +      8*(lane>>4) + i   (one 16B load)
//   frag elems 8..15 = K = kbase + 16 + 8*(lane>>4) + i   (16B load at +32B)
__device__ __forceinline__ Frag load_afrag(const unsigned short* rowBase /* + 8*(lane>>4) applied */,
                                           int kbase){
  Frag f;
  const uint4* p = (const uint4*)(rowBase + kbase);
  f.q[0] = p[0];
  f.q[1] = p[2];
  return f;
}

// ---------------------------------------------------------------------------
// fp32 -> bf16 row-major convert (one-time, for WMMA A operands)
// ---------------------------------------------------------------------------
__global__ void cvt_bf16_kernel(const float* __restrict__ src,
                                unsigned short* __restrict__ dst, int n){
  for (int i = blockIdx.x*blockDim.x + threadIdx.x; i < n; i += gridDim.x*blockDim.x)
    dst[i] = f2bf(src[i]);
}

// ---------------------------------------------------------------------------
// Pack a row-major fp32 weight [K][N] (or transpose [N][K]) into
// WMMA B-fragment-ordered bf16: idx = ((ntile*nk + kstep)*32 + lane)*16 + i
// holds W[kstep*32 + kmap(i,lane)][ntile*16 + (lane&15)],
// kmap(i,l) = ((i<8)? i : i+8) + 8*(l>>4)
// ---------------------------------------------------------------------------
__global__ void pack_b_kernel(const float* __restrict__ W, int ldw,
                              int K, int Nlog, int Npad, int transpose,
                              unsigned short* __restrict__ out){
  int nk = K >> 5;
  int total = (Npad >> 4) * nk * 512;
  for (int idx = blockIdx.x*blockDim.x + threadIdx.x; idx < total;
       idx += gridDim.x*blockDim.x){
    int i    = idx & 15;
    int lane = (idx >> 4) & 31;
    int fk   = idx >> 9;
    int kstep = fk % nk;
    int ntile = fk / nk;
    int kk = (kstep << 5) + ((i < 8) ? i : (i + 8)) + ((lane >> 4) << 3);
    int n  = (ntile << 4) + (lane & 15);
    float v = 0.f;
    if (n < Nlog)
      v = transpose ? W[(size_t)n * ldw + kk] : W[(size_t)kk * ldw + n];
    out[idx] = f2bf(v);
  }
}

// ---------------------------------------------------------------------------
// Generic WMMA GEMM: C[M,Nlog] = A[M,K] (bf16 row-major) @ Bpacked + bias
// Inner loop: 2x b128 A loads + 2x b128 B loads + 1 wmma. No conversion VALU.
// ---------------------------------------------------------------------------
__global__ void gemm_wmma_kernel(const unsigned short* __restrict__ A, int lda,
                                 const unsigned short* __restrict__ Bp,
                                 const float* __restrict__ bias,
                                 float* __restrict__ C, int ldc,
                                 int M, int Nlog, int Npad, int K){
  __builtin_prefetch(Bp, 0, 3);                     // global_prefetch_b8
  int wave = threadIdx.x >> 5, lane = threadIdx.x & 31;
  int mtile = blockIdx.x * (blockDim.x >> 5) + wave;
  int mtiles = (M + 15) >> 4;
  if (mtile >= mtiles) return;
  int row0 = mtile << 4;
  int arow = row0 + (lane & 15); if (arow >= M) arow = M - 1;
  const unsigned short* Ar = A + (size_t)arow * lda + ((lane >> 4) << 3);
  int nk = K >> 5, nt = Npad >> 4;
  for (int n = 0; n < nt; ++n){
    v8f acc = {};
    for (int k = 0; k < nk; ++k){
      Frag a = load_afrag(Ar, k << 5);
      Frag b;
      const uint4* bp = (const uint4*)(Bp + ((((size_t)n * nk + k) * 32 + lane) << 4));
      b.q[0] = bp[0]; b.q[1] = bp[1];
      acc = wmma_bf16(a, b, acc);
    }
    int col = (n << 4) + (lane & 15);
    if (col < Nlog){
      float bv = bias ? bias[col] : 0.f;
      #pragma unroll
      for (int v = 0; v < 8; ++v){
        int m = row0 + v + ((lane >> 4) << 3);
        if (m < M) C[(size_t)m * ldc + col] = acc[v] + bv;
      }
    }
  }
}

// ---------------------------------------------------------------------------
// Path attention pass 1: cw[b,s,l] = att( tanh( pt(full[b,s,l,:]) ) )
// full = [En[x0] | En[x2] | Ep[x1]] (384 wide) gathered as bf16 A fragments
// (two b128 loads each) from bf16 shadow tables. 1 block = (b,s); 8 waves = L
// tiles. The att weight vector is staged into LDS via the Tensor Data Mover.
// ---------------------------------------------------------------------------
__global__ void path_cw_kernel(const int* __restrict__ x,
                               const unsigned short* __restrict__ bEn,
                               const unsigned short* __restrict__ bEp,
                               const unsigned short* __restrict__ WptP,
                               const float* __restrict__ bpt,
                               const float* __restrict__ Watt,
                               const float* __restrict__ attB,
                               float* __restrict__ cw){
  __shared__ float s_watt[384];
#if __has_builtin(__builtin_amdgcn_tensor_load_to_lds) && __has_builtin(__builtin_amdgcn_s_wait_tensorcnt)
  if (threadIdx.x < 32){
    unsigned long long ga = (unsigned long long)(uintptr_t)Watt;
    unsigned int lds = (unsigned int)(uintptr_t)s_watt;
    // D# group0: count=1, lds_addr, global_addr[56:0], type=2
    u32x4 g0 = { 1u, lds, (unsigned int)ga,
                 (unsigned int)((ga >> 32) & 0x1FFFFFFu) | 0x80000000u };
    // D# group1: data_size=4B, tensor_dim0=384, tensor_dim1=1,
    //            tile_dim0=384, tile_dim1=1, tensor_dim0_stride=384
    i32x8 g1 = { (int)(2u << 16),
                 (int)((384u & 0xFFFFu) << 16),
                 (int)(1u << 16),
                 (int)(384u << 16),
                 1,
                 384,
                 0, 0 };
    i32x4 gz = {0,0,0,0};
#if defined(__clang_major__) && __clang_major__ >= 23
    i32x8 gz8 = {0,0,0,0,0,0,0,0};
    __builtin_amdgcn_tensor_load_to_lds(g0, g1, gz, gz, gz8, 0);
#else
    __builtin_amdgcn_tensor_load_to_lds(g0, g1, gz, gz, 0);
#endif
    __builtin_amdgcn_s_wait_tensorcnt(0);
  }
  __syncthreads();
#else
  for (int i = threadIdx.x; i < 384; i += blockDim.x) s_watt[i] = Watt[i];
  __syncthreads();
#endif

  int bs = blockIdx.x;                 // b*SS + s
  int wave = threadIdx.x >> 5, lane = threadIdx.x & 31;
  int row = (wave << 4) + (lane & 15); // path index within L
  const int* xr = x + ((size_t)bs * LLL + row) * 3;
  int x0 = xr[0], x1 = xr[1], x2 = xr[2];
  int koff = (lane >> 4) << 3;
  const unsigned short* seg0 = bEn + (size_t)x0 * EE + koff;
  const unsigned short* seg1 = bEn + (size_t)x2 * EE + koff;
  const unsigned short* seg2 = bEp + (size_t)x1 * EE + koff;

  // preload the 12 A fragments (16x32 bf16 each) for this wave's 16 rows
  Frag af[12];
  #pragma unroll
  for (int k = 0; k < 12; ++k){
    const unsigned short* base = ((k >> 2) == 0) ? seg0 : (((k >> 2) == 1) ? seg1 : seg2);
    af[k] = load_afrag(base, (k & 3) << 5);
  }

  float partial[8] = {0,0,0,0,0,0,0,0};
  for (int n = 0; n < 24; ++n){
    v8f acc = {};
    #pragma unroll
    for (int k = 0; k < 12; ++k){
      Frag b;
      const uint4* bp = (const uint4*)(WptP + ((((size_t)n * 12 + k) * 32 + lane) << 4));
      b.q[0] = bp[0]; b.q[1] = bp[1];
      acc = wmma_bf16(af[k], b, acc);
    }
    int col = (n << 4) + (lane & 15);
    float bb = bpt[col];
    float wa = s_watt[col];
    #pragma unroll
    for (int v = 0; v < 8; ++v) partial[v] += tanhf(acc[v] + bb) * wa;
  }
  // reduce across the 16 lanes that hold the same rows (different N columns)
  #pragma unroll
  for (int off = 8; off > 0; off >>= 1){
    #pragma unroll
    for (int v = 0; v < 8; ++v) partial[v] += __shfl_xor(partial[v], off, 16);
  }
  if ((lane & 15) == 0){
    float battv = attB[0];
    int mbase = (wave << 4) + ((lane >> 4) << 3);
    #pragma unroll
    for (int v = 0; v < 8; ++v)
      cw[(size_t)bs * LLL + mbase + v] = partial[v] + battv;
  }
}

// softmax over SEQ dim (dim=1), per (b,l): stats pass
__global__ void softmax_stats_kernel(const float* __restrict__ cw,
                                     float* __restrict__ mx, float* __restrict__ Z){
  int idx = blockIdx.x * (blockDim.x >> 5) + (threadIdx.x >> 5);  // b*LLL + l
  if (idx >= BB * LLL) return;
  int b = idx / LLL, l = idx % LLL;
  int lane = threadIdx.x & 31;
  const float* p = cw + ((size_t)b * SS) * LLL + l;
  float m = -3.0e38f;
  for (int s = lane; s < SS; s += 32) m = fmaxf(m, p[(size_t)s * LLL]);
  #pragma unroll
  for (int off = 16; off > 0; off >>= 1) m = fmaxf(m, __shfl_xor(m, off, 32));
  float z = 0.f;
  for (int s = lane; s < SS; s += 32) z += __expf(p[(size_t)s * LLL] - m);
  #pragma unroll
  for (int off = 16; off > 0; off >>= 1) z += __shfl_xor(z, off, 32);
  if (lane == 0){ mx[idx] = m; Z[idx] = z; }
}

// code_vectors[b,s,:] = sum_l full[b,s,l,:] * aw[b,s,l]   (fp32 tables, L2-resident)
__global__ void code_vectors_kernel(const int* __restrict__ x,
                                    const float* __restrict__ En,
                                    const float* __restrict__ Ep,
                                    const float* __restrict__ cw,
                                    const float* __restrict__ mx,
                                    const float* __restrict__ Z,
                                    float* __restrict__ cv){
  __shared__ float aw[LLL];
  __shared__ int xi0[LLL], xi1[LLL], xi2[LLL];
  int bs = blockIdx.x;
  int b = bs / SS;
  for (int l = threadIdx.x; l < LLL; l += blockDim.x){
    int id = b * LLL + l;
    aw[l] = __expf(cw[(size_t)bs * LLL + l] - mx[id]) / Z[id];
    const int* xr = x + ((size_t)bs * LLL + l) * 3;
    xi0[l] = xr[0]; xi1[l] = xr[1]; xi2[l] = xr[2];
  }
  __syncthreads();
  for (int c = threadIdx.x; c < 3 * EE; c += blockDim.x){
    int segI = c >> 7, cc = c & 127;
    float acc = 0.f;
    for (int l = 0; l < LLL; ++l){
      const float* base = (segI == 0) ? En + (size_t)xi0[l] * EE
                        : (segI == 1) ? En + (size_t)xi2[l] * EE
                                      : Ep + (size_t)xi1[l] * EE;
      acc += aw[l] * base[cc];
    }
    cv[(size_t)bs * (3 * EE) + c] = acc;
  }
}

// cumulative-sigmoid error pooling over transformed table G (bias folded in)
__global__ void err_pool_kernel(const int* __restrict__ errIdx,
                                const float* __restrict__ G,
                                float* __restrict__ pooled){
  int b = blockIdx.x / TTT, t = blockIdx.x % TTT;
  int e = threadIdx.x;
  float th = G[ERRTH * EE + e];
  float acc = 0.f;
  for (int s = 0; s < SS; ++s){
    int id = errIdx[((size_t)b * SS + s) * TTT + t];
    acc += G[(size_t)id * EE + e];
    float v = sigf(acc - th);
    atomicAdd(&pooled[((size_t)b * SS + s) * EE + e], v * (1.0f / TTT));
  }
}

// LSTM scan: one workgroup, 4 waves; WMMA h @ Whh^T each step. bf16 shadow of
// h in LDS -> A fragments are two ds_load_b128, no per-step conversion chains.
__global__ void lstm_scan_kernel(const float* __restrict__ X,   // [B*S,512] = x@Wih^T + bih + bhh
                                 const unsigned short* __restrict__ WhhP,
                                 float* __restrict__ h_out){
  __shared__ float cbuf[BB * EE];
  __shared__ unsigned short hbf[BB * EE];      // bf16 shadow of h
  __shared__ float gbuf[BB * 512];
  int t = threadIdx.x;                 // 128 threads
  int wave = t >> 5, lane = t & 31;
  for (int i = t; i < BB * EE; i += 128){ cbuf[i] = 0.f; hbf[i] = 0; }
  __syncthreads();
  for (int s = 0; s < SS; ++s){
    Frag af[4];
    int brow = lane & 15, koff = (lane >> 4) << 3;
    const unsigned short* hrow = hbf + brow * EE + koff;
    #pragma unroll
    for (int k = 0; k < 4; ++k) af[k] = load_afrag(hrow, k << 5);
    #pragma unroll
    for (int nn = 0; nn < 8; ++nn){
      int n = wave * 8 + nn;
      v8f acc = {};
      #pragma unroll
      for (int k = 0; k < 4; ++k){
        Frag b;
        const uint4* bp = (const uint4*)(WhhP + ((((size_t)n * 4 + k) * 32 + lane) << 4));
        b.q[0] = bp[0]; b.q[1] = bp[1];
        acc = wmma_bf16(af[k], b, acc);
      }
      int col = (n << 4) + (lane & 15);
      #pragma unroll
      for (int v = 0; v < 8; ++v){
        int bi = v + ((lane >> 4) << 3);
        gbuf[bi * 512 + col] = acc[v] + X[((size_t)bi * SS + s) * 512 + col];
      }
    }
    __syncthreads();
    for (int i = t; i < BB * EE; i += 128){
      int bi = i >> 7, e = i & 127;
      float gi = gbuf[bi * 512 + e];
      float gf = gbuf[bi * 512 + 128 + e];
      float gg = gbuf[bi * 512 + 256 + e];
      float go = gbuf[bi * 512 + 384 + e];
      float c = sigf(gf) * cbuf[i] + sigf(gi) * tanhf(gg);
      float h = sigf(go) * tanhf(c);
      cbuf[i] = c; hbf[i] = f2bf(h);
      h_out[((size_t)bi * SS + s) * EE + e] = h;
    }
    __syncthreads();
  }
}

// ------------------------------ elementwise helpers -------------------------
__global__ void zero_kernel(float* p, int n){
  for (int i = blockIdx.x*blockDim.x + threadIdx.x; i < n; i += gridDim.x*blockDim.x) p[i] = 0.f;
}
__global__ void gather_rows_kernel(const float* __restrict__ tab, const int* __restrict__ idx,
                                   float* __restrict__ dst, int rows){
  for (int i = blockIdx.x*blockDim.x + threadIdx.x; i < rows * EE; i += gridDim.x*blockDim.x){
    int r = i >> 7, c = i & 127;
    dst[i] = tab[(size_t)idx[r] * EE + c];
  }
}
// concat of 3 fp32 sources, optional fp32 and/or bf16 destinations
__global__ void concat3_dual_kernel(const float* s0, int w0, const float* s1, int w1,
                                    const float* s2, int w2,
                                    float* dstF, unsigned short* dstB, int rows){
  int W = w0 + w1 + w2;
  for (int i = blockIdx.x*blockDim.x + threadIdx.x; i < rows * W; i += gridDim.x*blockDim.x){
    int r = i / W, c = i % W;
    float v;
    if (c < w0)            v = s0[(size_t)r * w0 + c];
    else if (c < w0 + w1)  v = s1[(size_t)r * w1 + (c - w0)];
    else                   v = s2[(size_t)r * w2 + (c - w0 - w1)];
    if (dstF) dstF[i] = v;
    if (dstB) dstB[i] = f2bf(v);
  }
}
__global__ void bias_comb_kernel(const float* a, const float* b, float* o, int n){
  for (int i = blockIdx.x*blockDim.x + threadIdx.x; i < n; i += gridDim.x*blockDim.x) o[i] = a[i] + b[i];
}
__global__ void gains_kernel(const float* i1o, const float* i2o, const float* i3o, const float* i4o,
                             const float* ia1, const float* ia2,
                             float* cg, float* rg, float* intra, int n){
  float A1 = ia1[0], A2 = ia2[0];
  for (int i = blockIdx.x*blockDim.x + threadIdx.x; i < n; i += gridDim.x*blockDim.x){
    float c = tanhf(i1o[i]) * sigf(i2o[i]);
    float r = tanhf(i3o[i]) * sigf(i4o[i]);
    cg[i] = c; rg[i] = r; intra[i] = A1 * c + A2 * r;
  }
}
__global__ void errh_delta_kernel(const float* h, const float* intra,
                                  const float* a1, const float* a2,
                                  float* err_h, float* delta, int n){
  float A1 = a1[0], A2 = a2[0];
  for (int i = blockIdx.x*blockDim.x + threadIdx.x; i < n; i += gridDim.x*blockDim.x){
    float e = A1 * h[i] + A2 * intra[i];
    err_h[i] = e;
    int r = i >> 7;            // (b*S + s)
    int s = r & (SS - 1);
    float d = 0.f;
    if (s != 0) d = e - (A1 * h[i - EE] + A2 * intra[i - EE]);
    delta[i] = d;
  }
}
// d = diff_input - err_h, emitted directly as bf16 (feeds ds1/ds2 WMMA GEMMs)
__global__ void sub_d_bf_kernel(const float* di, const float* eh, unsigned short* d, int n){
  for (int i = blockIdx.x*blockDim.x + threadIdx.x; i < n; i += gridDim.x*blockDim.x)
    d[i] = f2bf(di[i] - eh[i]);
}
// sub_h = a3*err_h + a4*tanh(ds1)*sig(ds2); fp32 output + bf16 copy (feeds fc)
__global__ void subh_kernel(const float* d1, const float* d2, const float* eh,
                            const float* a3, const float* a4,
                            float* sh, unsigned short* shb, int n){
  float A3 = a3[0], A4 = a4[0];
  for (int i = blockIdx.x*blockDim.x + threadIdx.x; i < n; i += gridDim.x*blockDim.x){
    float sa = tanhf(d1[i]) * sigf(d2[i]);
    float v = A3 * eh[i] + A4 * sa;
    sh[i] = v; shb[i] = f2bf(v);
  }
}
__global__ void sigmoid_kernel(float* p, int n){
  for (int i = blockIdx.x*blockDim.x + threadIdx.x; i < n; i += gridDim.x*blockDim.x) p[i] = sigf(p[i]);
}

// ---------------------------------------------------------------------------
// Launch orchestration
// ---------------------------------------------------------------------------
extern "C" void kernel_launch(void* const* d_in, const int* in_sizes, int n_in,
                              void* d_out, int out_size, void* d_ws, size_t ws_size,
                              hipStream_t stream){
  (void)in_sizes; (void)n_in; (void)out_size; (void)ws_size;
  const int*   x     = (const int*)d_in[0];
  const int*   xqa   = (const int*)d_in[1];
  const int*   xq    = (const int*)d_in[2];
  const int*   xerr  = (const int*)d_in[3];
  const int*   xrank = (const int*)d_in[4];
  const int*   xdiff = (const int*)d_in[5];
  const float* En    = (const float*)d_in[6];
  const float* Ep    = (const float*)d_in[7];
  const float* Eqa   = (const float*)d_in[8];
  const float* Eq    = (const float*)d_in[9];
  const float* Eerr  = (const float*)d_in[10];
  const float* Erank = (const float*)d_in[11];
  const float* Ediff = (const float*)d_in[12];
  const float* qaW   = (const float*)d_in[13]; const float* qaB  = (const float*)d_in[14];
  const float* qW    = (const float*)d_in[15]; const float* qB   = (const float*)d_in[16];
  const float* errW  = (const float*)d_in[17]; const float* errB = (const float*)d_in[18];
  const float* rankW = (const float*)d_in[19]; const float* rankB= (const float*)d_in[20];
  const float* diffW = (const float*)d_in[21]; const float* diffB= (const float*)d_in[22];
  const float* dtW   = (const float*)d_in[23]; const float* dtB  = (const float*)d_in[24];
  const float* i1W   = (const float*)d_in[25]; const float* i1B  = (const float*)d_in[26];
  const float* i2W   = (const float*)d_in[27]; const float* i2B  = (const float*)d_in[28];
  const float* i3W   = (const float*)d_in[29]; const float* i3B  = (const float*)d_in[30];
  const float* i4W   = (const float*)d_in[31]; const float* i4B  = (const float*)d_in[32];
  const float* ptW   = (const float*)d_in[33]; const float* ptB  = (const float*)d_in[34];
  const float* attW  = (const float*)d_in[35]; const float* attB = (const float*)d_in[36];
  const float* ds1W  = (const float*)d_in[37]; const float* ds1B = (const float*)d_in[38];
  const float* ds2W  = (const float*)d_in[39]; const float* ds2B = (const float*)d_in[40];
  const float* fcW   = (const float*)d_in[41]; const float* fcB  = (const float*)d_in[42];
  const float* Wih   = (const float*)d_in[43];
  const float* Whh   = (const float*)d_in[44];
  const float* bih   = (const float*)d_in[45];
  const float* bhh   = (const float*)d_in[46];
  const float* ia1   = (const float*)d_in[47]; const float* ia2  = (const float*)d_in[48];
  const float* a1    = (const float*)d_in[49]; const float* a2   = (const float*)d_in[50];
  const float* a3    = (const float*)d_in[51]; const float* a4   = (const float*)d_in[52];

  // ---- output layout (floats), reference tuple order ----
  float* out = (float*)d_out;
  float* o_res   = out + 0;         // [B,S,100]
  float* o_h     = out + 204800;    // [B,S,E]
  float* o_errh  = out + 466944;
  float* o_subh  = out + 729088;
  float* o_qa    = out + 991232;
  float* o_cv    = out + 1253376;   // [B,S,3E]
  float* o_rnn   = out + 2039808;   // [B,S,4E]
  float* o_diffe = out + 3088384;
  float* o_delta = out + 3350528;
  float* o_cg    = out + 3612672;
  float* o_rg    = out + 3874816;

  // ---- workspace layout: fp32 region ----
  float* W = (float*)d_ws;
  float* wEqa   = W + 0;        // 201*128 (transformed tables, gather sources)
  float* wEq    = W + 25728;    // 101*128
  float* wErank = W + 38656;    // 11*128
  float* wEdiff = W + 40064;    // 150*128
  float* wEerr  = W + 59264;    // 100*128
  float* wQemb  = W + 72064;    // 2048*128
  float* wRank  = W + 334208;   // 2048*128
  float* wPool  = W + 596352;   // 2048*128
  float* wCw    = W + 858496;   // 2048*128  (cw[b,s,l])
  float* wMx    = W + 1120640;  // 2048
  float* wZ     = W + 1122688;  // 2048
  float* wI1    = W + 1124736;  // 2048*128
  float* wI2    = W + 1386880;
  float* wI3    = W + 1649024;
  float* wI4    = W + 1911168;
  float* wIntra = W + 2173312;
  float* wX     = W + 2435456;  // 2048*512
  float* wDin   = W + 3484032;  // 2048*128
  float* wBcomb = W + 3746176;  // 512
  // ---- workspace layout: bf16 region (all offsets multiple of 8 => 16B aligned)
  unsigned short* U = (unsigned short*)(W + 3746688);
  unsigned short* bEn   = U + 0;        // 10002*128
  unsigned short* bEp   = U + 1280256;  // 10002*128
  unsigned short* bEqa  = U + 2560512;  // 201*128
  unsigned short* bEq   = U + 2586240;  // 101*128
  unsigned short* bErank= U + 2599168;  // 11*128
  unsigned short* bEdiff= U + 2600576;  // 150*128
  unsigned short* bEerr = U + 2619776;  // 100*128
  unsigned short* bCin  = U + 2632576;  // 2048*640
  unsigned short* bRin  = U + 3943296;  // 2048*640
  unsigned short* bRnn  = U + 5254016;  // 2048*512
  unsigned short* bDx   = U + 6302592;  // 2048*384
  unsigned short* bD    = U + 7089024;  // 2048*128
  unsigned short* bSubh = U + 7351168;  // 2048*128
  // ---- packed bf16 B-fragment weights ----
  unsigned short* P = U + 7613312;
  unsigned short* pQA  = P + 0;        // ExE -> 16384 each
  unsigned short* pQ   = P + 16384;
  unsigned short* pRK  = P + 32768;
  unsigned short* pDF  = P + 49152;
  unsigned short* pER  = P + 65536;
  unsigned short* pI1  = P + 81920;    // 640->128 : 81920 each
  unsigned short* pI2  = P + 163840;
  unsigned short* pI3  = P + 245760;
  unsigned short* pI4  = P + 327680;
  unsigned short* pDT  = P + 409600;   // 384->128 : 49152
  unsigned short* pPT  = P + 458752;   // 384->384 : 147456
  unsigned short* pDS1 = P + 606208;
  unsigned short* pDS2 = P + 622592;
  unsigned short* pFC  = P + 638976;   // 128->112pad : 14336
  unsigned short* pWIH = P + 653312;   // 512->512 : 262144
  unsigned short* pWHH = P + 915456;   // 128->512 : 65536

  dim3 blk(256);
  dim3 gE(256);                         // grid-stride elementwise
  #define PACK(src, ldw, K, N, Npad, tr, dst) \
    pack_b_kernel<<<dim3((((Npad)>>4)*((K)>>5)*512 + 255)/256), blk, 0, stream>>>(src, ldw, K, N, Npad, tr, dst)

  // 1) pack all weights into WMMA B-fragment order (bf16)
  PACK(qaW, 128, 128, 128, 128, 0, pQA);
  PACK(qW , 128, 128, 128, 128, 0, pQ );
  PACK(rankW,128,128, 128, 128, 0, pRK);
  PACK(diffW,128,128, 128, 128, 0, pDF);
  PACK(errW, 128,128, 128, 128, 0, pER);
  PACK(i1W, 128, 640, 128, 128, 0, pI1);
  PACK(i2W, 128, 640, 128, 128, 0, pI2);
  PACK(i3W, 128, 640, 128, 128, 0, pI3);
  PACK(i4W, 128, 640, 128, 128, 0, pI4);
  PACK(dtW, 128, 384, 128, 128, 0, pDT);
  PACK(ptW, 384, 384, 384, 384, 0, pPT);
  PACK(ds1W,128, 128, 128, 128, 0, pDS1);
  PACK(ds2W,128, 128, 128, 128, 0, pDS2);
  PACK(fcW, 100, 128, 100, 112, 0, pFC);
  PACK(Wih, 512, 512, 512, 512, 1, pWIH);   // x @ Wih^T
  PACK(Whh, 128, 128, 512, 512, 1, pWHH);   // h @ Whh^T

  bias_comb_kernel<<<gE, blk, 0, stream>>>(bih, bhh, wBcomb, 512);

  // 2) bf16 shadows of A-side tables (one-time)
  cvt_bf16_kernel<<<gE, blk, 0, stream>>>(En,   bEn,   10002*EE);
  cvt_bf16_kernel<<<gE, blk, 0, stream>>>(Ep,   bEp,   10002*EE);
  cvt_bf16_kernel<<<gE, blk, 0, stream>>>(Eqa,  bEqa,  201*EE);
  cvt_bf16_kernel<<<gE, blk, 0, stream>>>(Eq,   bEq,   101*EE);
  cvt_bf16_kernel<<<gE, blk, 0, stream>>>(Erank,bErank,11*EE);
  cvt_bf16_kernel<<<gE, blk, 0, stream>>>(Ediff,bEdiff,150*EE);
  cvt_bf16_kernel<<<gE, blk, 0, stream>>>(Eerr, bEerr, 100*EE);

  // 3) fold linear layers into embedding tables (tiny WMMA GEMMs): E' = E @ W + b
  #define TGEMM(A, Bp, bias, C, M) \
    gemm_wmma_kernel<<<dim3(((((M)+15)>>4)+7)/8), blk, 0, stream>>>(A, 128, Bp, bias, C, 128, M, 128, 128, 128)
  TGEMM(bEqa,  pQA, qaB,  wEqa,  201);
  TGEMM(bEq,   pQ,  qB,   wEq,   101);
  TGEMM(bErank,pRK, rankB,wErank, 11);
  TGEMM(bEdiff,pDF, diffB,wEdiff,150);
  TGEMM(bEerr, pER, errB, wEerr, 100);

  // 4) gathers (embedding + linear collapsed to one lookup)
  gather_rows_kernel<<<gE, blk, 0, stream>>>(wEqa,  xqa,  o_qa,   ROWS);
  gather_rows_kernel<<<gE, blk, 0, stream>>>(wEq,   xq,   wQemb,  ROWS);
  gather_rows_kernel<<<gE, blk, 0, stream>>>(wErank,xrank,wRank,  ROWS);
  gather_rows_kernel<<<gE, blk, 0, stream>>>(wEdiff,xdiff,o_diffe,ROWS);

  // 5) cumulative-sigmoid error pooling
  zero_kernel<<<gE, blk, 0, stream>>>(wPool, ROWS * EE);
  err_pool_kernel<<<dim3(BB * TTT), dim3(128), 0, stream>>>(xerr, wEerr, wPool);

  // 6) code2vec path attention
  path_cw_kernel<<<dim3(ROWS), blk, 0, stream>>>(x, bEn, bEp, pPT, ptB, attW, attB, wCw);
  softmax_stats_kernel<<<dim3(256), blk, 0, stream>>>(wCw, wMx, wZ);
  code_vectors_kernel<<<dim3(ROWS), blk, 0, stream>>>(x, En, Ep, wCw, wMx, wZ, o_cv);

  // 7) intra-error gated units (concat inputs emitted directly as bf16)
  concat3_dual_kernel<<<gE, blk, 0, stream>>>(o_qa, 128, o_cv, 384, wPool, 128,
                                              (float*)0, bCin, ROWS);
  concat3_dual_kernel<<<gE, blk, 0, stream>>>(o_qa, 128, o_cv, 384, wRank, 128,
                                              (float*)0, bRin, ROWS);
  gemm_wmma_kernel<<<dim3(16), blk, 0, stream>>>(bCin, 640, pI1, i1B, wI1, 128, ROWS, 128, 128, 640);
  gemm_wmma_kernel<<<dim3(16), blk, 0, stream>>>(bCin, 640, pI2, i2B, wI2, 128, ROWS, 128, 128, 640);
  gemm_wmma_kernel<<<dim3(16), blk, 0, stream>>>(bRin, 640, pI3, i3B, wI3, 128, ROWS, 128, 128, 640);
  gemm_wmma_kernel<<<dim3(16), blk, 0, stream>>>(bRin, 640, pI4, i4B, wI4, 128, ROWS, 128, 128, 640);
  gains_kernel<<<gE, blk, 0, stream>>>(wI1, wI2, wI3, wI4, ia1, ia2, o_cg, o_rg, wIntra, ROWS * EE);

  // 8) LSTM: hoisted input projection + sequential recurrence
  concat3_dual_kernel<<<gE, blk, 0, stream>>>(o_qa, 128, o_cv, 384, (const float*)0, 0,
                                              o_rnn, bRnn, ROWS);
  gemm_wmma_kernel<<<dim3(16), blk, 0, stream>>>(bRnn, 512, pWIH, wBcomb, wX, 512, ROWS, 512, 512, 512);
  lstm_scan_kernel<<<dim3(1), dim3(128), 0, stream>>>(wX, pWHH, o_h);

  // 9) err_h / delta
  errh_delta_kernel<<<gE, blk, 0, stream>>>(o_h, wIntra, a1, a2, o_errh, o_delta, ROWS * EE);

  // 10) diff transform + sub acquisition + head
  concat3_dual_kernel<<<gE, blk, 0, stream>>>(wQemb, 128, o_diffe, 128, o_delta, 128,
                                              (float*)0, bDx, ROWS);
  gemm_wmma_kernel<<<dim3(16), blk, 0, stream>>>(bDx, 384, pDT, dtB, wDin, 128, ROWS, 128, 128, 384);
  sub_d_bf_kernel<<<gE, blk, 0, stream>>>(wDin, o_errh, bD, ROWS * EE);
  gemm_wmma_kernel<<<dim3(16), blk, 0, stream>>>(bD, 128, pDS1, ds1B, wI1, 128, ROWS, 128, 128, 128);
  gemm_wmma_kernel<<<dim3(16), blk, 0, stream>>>(bD, 128, pDS2, ds2B, wI2, 128, ROWS, 128, 128, 128);
  subh_kernel<<<gE, blk, 0, stream>>>(wI1, wI2, o_errh, a3, a4, o_subh, bSubh, ROWS * EE);
  gemm_wmma_kernel<<<dim3(16), blk, 0, stream>>>(bSubh, 128, pFC, fcB, o_res, 100, ROWS, 100, 112, 128);
  sigmoid_kernel<<<gE, blk, 0, stream>>>(o_res, ROWS * OUTD);
  #undef PACK
  #undef TGEMM
}
